// QuantumFluxModel_4097398800531
// MI455X (gfx1250) — compile-verified, matrix operand order checked by reading, not run on hardware
//
#include <hip/hip_runtime.h>
#include <hip/hip_bf16.h>

// ---------------------------------------------------------------------------
// QuantumFluxModel on MI455X (gfx1250): bf16 WMMA GEMMs + fp32 epilogues.
// ---------------------------------------------------------------------------
#define QB 2
#define QS 2048
#define QD 1024
#define QL 8
#define QV 32000
#define QDT 0.055f           // 0.5*(dt_min+dt_max)
#define QBOOST 1.55f         // 1 + dt/dt_max
#define QSCALE 0.03125f      // 1/sqrt(D)

typedef __bf16 bf16_t;
typedef __attribute__((ext_vector_type(16))) __bf16 v16bf;
typedef __attribute__((ext_vector_type(8)))  __bf16 v8bf;
typedef __attribute__((ext_vector_type(8)))  float  v8f;

// ---------------------------------------------------------------------------
// WMMA helpers (CDNA5 16x16x32 bf16 -> f32)
// ---------------------------------------------------------------------------
__device__ __forceinline__ v8f qfm_wmma(v16bf a, v16bf b, v8f c) {
  return __builtin_amdgcn_wmma_f32_16x16x32_bf16(false, a, false, b, (short)0, c,
                                                 false, false);
}

// Fragment from a row-major bf16 matrix (global or LDS pointer).
// Lane l holds row (lane&15); half=(lane>>4): elements 0..7 = K k0+8*half..+7,
// elements 8..15 = K k0+16+8*half..+7  (per ISA 16-bit 16x32 fragment table).
__device__ __forceinline__ v16bf qfm_frag_row(const __hip_bfloat16* rowbase,
                                              int k0, int half) {
  const __hip_bfloat16* q = rowbase + k0 + half * 8;
  v8bf lo = *(const v8bf*)(q);
  v8bf hi = *(const v8bf*)(q + 16);
  v16bf f;
#pragma unroll
  for (int i = 0; i < 8; ++i) { f[i] = lo[i]; f[i + 8] = hi[i]; }
  return f;
}

// ---------------------------------------------------------------------------
// GEMM-NT with causal mask: score = tril(A[b] * A[b]^T) * scale, bf16 out.
// Block = 128 threads (4 waves); block tile 128x64; wave tile 32x64
// (8 accumulators, B fragments reused across both M-subtiles).
// ---------------------------------------------------------------------------
__global__ void __launch_bounds__(128)
qfm_gemm_nt_causal(const __hip_bfloat16* __restrict__ A,
                   __hip_bfloat16* __restrict__ Cb,
                   int S, int K, float scale) {
  const int b = blockIdx.z;
  const __hip_bfloat16* Ab = A + (size_t)b * S * K;
  __hip_bfloat16* Cbb = Cb + (size_t)b * S * S;
  const int m0 = blockIdx.x * 128;
  const int n0 = blockIdx.y * 64;
  const int tid = threadIdx.x;

  if (n0 > m0 + 127) {                // tile entirely above diagonal -> zeros
    for (int i = tid; i < 128 * 64; i += 128) {
      int r = i >> 6, c = i & 63;
      Cbb[(size_t)(m0 + r) * S + (n0 + c)] = __float2bfloat16(0.f);
    }
    return;
  }

  const int wave = tid >> 5, lane = tid & 31;
  const int half = lane >> 4, lm = lane & 15;
  const int mrow = m0 + wave * 32;

  v8f acc[2][4] = {{{}, {}, {}, {}}, {{}, {}, {}, {}}};
  const __hip_bfloat16* arow0 = Ab + (size_t)(mrow + lm) * K;
  const __hip_bfloat16* arow1 = Ab + (size_t)(mrow + 16 + lm) * K;
  const __hip_bfloat16* brow[4];
#pragma unroll
  for (int t = 0; t < 4; ++t) brow[t] = Ab + (size_t)(n0 + 16 * t + lm) * K;

  for (int k0 = 0; k0 < K; k0 += 32) {
    v16bf a0 = qfm_frag_row(arow0, k0, half);
    v16bf a1 = qfm_frag_row(arow1, k0, half);
#pragma unroll
    for (int t = 0; t < 4; ++t) {
      v16bf bf = qfm_frag_row(brow[t], k0, half);
      acc[0][t] = qfm_wmma(a0, bf, acc[0][t]);
      acc[1][t] = qfm_wmma(a1, bf, acc[1][t]);
    }
  }

#pragma unroll
  for (int s = 0; s < 2; ++s) {
#pragma unroll
    for (int t = 0; t < 4; ++t) {
      int n = n0 + 16 * t + lm;
#pragma unroll
      for (int v = 0; v < 8; ++v) {
        int m = mrow + 16 * s + v + 8 * half;
        float val = (n <= m) ? acc[s][t][v] * scale : 0.f;  // causal: keep k<=q
        Cbb[(size_t)m * S + n] = __float2bfloat16(val);
      }
    }
  }
}

// ---------------------------------------------------------------------------
// GEMM-NN: C[b] = A[b](MxK, row-major, lda=K) * B[b](KxN, row-major) (+bias).
// Block = 256 threads (8 waves); block tile 128x64.
// B tile staged TRANSPOSED through double-buffered LDS ([N=64][K=32], rows
// padded to 40 elems => conflict-free b128 ds reads). One barrier per K-step;
// next tile's global fetch overlaps current tile's 32 WMMAs.
// ---------------------------------------------------------------------------
#define QNN_PAD 40

__global__ void __launch_bounds__(256)
qfm_gemm_nn(const __hip_bfloat16* __restrict__ A,
            const __hip_bfloat16* __restrict__ Bm,
            float* __restrict__ C,
            const float* __restrict__ bias,
            int M, int N, int K, int ldb, int ldc,
            size_t sA, size_t sB, size_t sC) {
  const int bz = blockIdx.z;
  A += (size_t)bz * sA;  Bm += (size_t)bz * sB;  C += (size_t)bz * sC;
  const int m0 = blockIdx.x * 128;
  const int n0 = blockIdx.y * 64;
  const int tid = threadIdx.x;
  const int wave = tid >> 5, lane = tid & 31;
  const int half = lane >> 4, lm = lane & 15;

  __shared__ __align__(16) __hip_bfloat16 Bst[2][64 * QNN_PAD];  // [n][k] padded

  v8f acc[4] = {{}, {}, {}, {}};
  const __hip_bfloat16* arow = A + (size_t)(m0 + wave * 16 + lm) * K;
  const int lk = tid >> 3;            // 0..31  (K row of the global tile)
  const int lc = (tid & 7) * 8;       // 0..56  (N col of the global tile)
  const __hip_bfloat16* gsrc = Bm + (size_t)lk * ldb + n0 + lc;

  // prologue: stage tile 0
  v8bf regs = *(const v8bf*)(gsrc);
#pragma unroll
  for (int j = 0; j < 8; ++j)
    *(__bf16*)&Bst[0][(lc + j) * QNN_PAD + lk] = regs[j];
  __syncthreads();

  const int T = K >> 5;
  for (int i = 0; i < T; ++i) {
    const bool more = (i + 1 < T);
    if (more)                                   // fetch next tile early
      regs = *(const v8bf*)(gsrc + (size_t)(i + 1) * 32 * ldb);

    v16bf af = qfm_frag_row(arow, i << 5, half);
    const __hip_bfloat16* buf = Bst[i & 1];
#pragma unroll
    for (int t = 0; t < 4; ++t) {
      v16bf bf = qfm_frag_row(&buf[(16 * t + lm) * QNN_PAD], 0, half);
      acc[t] = qfm_wmma(af, bf, acc[t]);
    }

    if (more) {                                 // stage next tile, single barrier
#pragma unroll
      for (int j = 0; j < 8; ++j)
        *(__bf16*)&Bst[(i + 1) & 1][(lc + j) * QNN_PAD + lk] = regs[j];
      __syncthreads();
    }
  }

#pragma unroll
  for (int t = 0; t < 4; ++t) {
    int n = n0 + 16 * t + lm;
    float bv = bias ? bias[n] : 0.f;
#pragma unroll
    for (int v = 0; v < 8; ++v) {
      int m = m0 + wave * 16 + v + 8 * half;
      C[(size_t)m * ldc + n] = acc[t][v] + bv;
    }
  }
}

// ---------------------------------------------------------------------------
// Reductions
// ---------------------------------------------------------------------------
__device__ __forceinline__ float qfm_block_sum(float v, float* red) {
  int tid = threadIdx.x;
  red[tid] = v; __syncthreads();
  for (int s = blockDim.x >> 1; s > 0; s >>= 1) {
    if (tid < s) red[tid] += red[tid + s];
    __syncthreads();
  }
  float r = red[0]; __syncthreads();
  return r;
}
__device__ __forceinline__ float qfm_block_max(float v, float* red) {
  int tid = threadIdx.x;
  red[tid] = v; __syncthreads();
  for (int s = blockDim.x >> 1; s > 0; s >>= 1) {
    if (tid < s) red[tid] = fmaxf(red[tid], red[tid + s]);
    __syncthreads();
  }
  float r = red[0]; __syncthreads();
  return r;
}

// ---------------------------------------------------------------------------
// Elementwise / row kernels
// ---------------------------------------------------------------------------
__global__ void __launch_bounds__(256)
qfm_encode(const int* __restrict__ ids, float* __restrict__ st,
           __hip_bfloat16* __restrict__ stb) {
  int row = blockIdx.x;                       // B*S rows
  float theta = 6.28318530718f * (float)ids[row] / (float)QV;
  size_t base = (size_t)row * QD;
  for (int d = threadIdx.x; d < QD; d += blockDim.x) {
    bool is_cos = d < (QD / 2);
    float k = is_cos ? (float)(d + 1) : (float)(d - QD / 2 + 1);
    float ang = theta * k;
    float val = (is_cos ? cosf(ang) : sinf(ang)) * QSCALE;
    st[base + d] = val;
    stb[base + d] = __float2bfloat16(val);
  }
}

__global__ void __launch_bounds__(256)
qfm_cast_bf16(const float* __restrict__ x, __hip_bfloat16* __restrict__ y, size_t n) {
  for (size_t i = (size_t)blockIdx.x * blockDim.x + threadIdx.x; i < n;
       i += (size_t)gridDim.x * blockDim.x)
    y[i] = __float2bfloat16(x[i]);
}

__global__ void __launch_bounds__(256)
qfm_mid(const float* __restrict__ st, const float* __restrict__ k1,
        __hip_bfloat16* __restrict__ outb, size_t n) {
  for (size_t i = (size_t)blockIdx.x * blockDim.x + threadIdx.x; i < n;
       i += (size_t)gridDim.x * blockDim.x)
    outb[i] = __float2bfloat16(st[i] + QDT * k1[i]);
}

__global__ void __launch_bounds__(256)
qfm_heun(const float* __restrict__ st, float* __restrict__ k1,
         const float* __restrict__ k2, size_t n) {
  for (size_t i = (size_t)blockIdx.x * blockDim.x + threadIdx.x; i < n;
       i += (size_t)gridDim.x * blockDim.x)
    k1[i] = st[i] + 0.5f * QDT * (k1[i] + k2[i]);   // ev written over k1
}

__global__ void __launch_bounds__(256)
qfm_energy(const float* __restrict__ ev, float* __restrict__ en) {
  __shared__ float red[256];
  int row = blockIdx.x;
  const float* r = ev + (size_t)row * QD;
  float s = 0.f;
  for (int d = threadIdx.x; d < QD; d += blockDim.x) { float t = r[d]; s += t * t; }
  float tot = qfm_block_sum(s, red);
  if (threadIdx.x == 0) en[row] = sqrtf(tot);
}

__global__ void __launch_bounds__(256)
qfm_thr(const float* __restrict__ en, float* __restrict__ thr) {
  __shared__ float red[256];
  int b = blockIdx.x;
  float s = 0.f;
  for (int i = threadIdx.x; i < QS; i += blockDim.x) s += en[b * QS + i];
  float tot = qfm_block_sum(s, red);
  if (threadIdx.x == 0) thr[b] = tot / (float)QS;
}

__global__ void __launch_bounds__(256)
qfm_tunnel(const float* __restrict__ ev, const float* __restrict__ en,
           const float* __restrict__ thr, __hip_bfloat16* __restrict__ outb) {
  int row = blockIdx.x;
  int b = row / QS;
  float f = (en[row] < thr[b]) ? QBOOST : 1.f;
  size_t base = (size_t)row * QD;
  for (int d = threadIdx.x; d < QD; d += blockDim.x)
    outb[base + d] = __float2bfloat16(ev[base + d] * f);
}

__global__ void __launch_bounds__(256)
qfm_ln(const float* __restrict__ y, const float* __restrict__ w,
       const float* __restrict__ b, float* __restrict__ outf,
       __hip_bfloat16* __restrict__ outb) {
  __shared__ float red[256];
  int row = blockIdx.x;
  const float* yr = y + (size_t)row * QD;
  float s = 0.f;
  for (int d = threadIdx.x; d < QD; d += blockDim.x) s += yr[d];
  float mu = qfm_block_sum(s, red) / (float)QD;
  float v = 0.f;
  for (int d = threadIdx.x; d < QD; d += blockDim.x) { float t = yr[d] - mu; v += t * t; }
  float rstd = rsqrtf(qfm_block_sum(v, red) / (float)QD + 1e-5f);
  size_t base = (size_t)row * QD;
  for (int d = threadIdx.x; d < QD; d += blockDim.x) {
    float o = (yr[d] - mu) * rstd * w[d] + b[d];
    outf[base + d] = o;
    outb[base + d] = __float2bfloat16(o);
  }
}

// final LN + energy normalization -> bf16 for logits GEMM
__global__ void __launch_bounds__(256)
qfm_final(const float* __restrict__ st, const float* __restrict__ w,
          const float* __restrict__ b, __hip_bfloat16* __restrict__ outb) {
  __shared__ float red[256];
  __shared__ float tmp[QD];
  int row = blockIdx.x;
  const float* yr = st + (size_t)row * QD;
  float s = 0.f;
  for (int d = threadIdx.x; d < QD; d += blockDim.x) s += yr[d];
  float mu = qfm_block_sum(s, red) / (float)QD;
  float v = 0.f;
  for (int d = threadIdx.x; d < QD; d += blockDim.x) { float t = yr[d] - mu; v += t * t; }
  float rstd = rsqrtf(qfm_block_sum(v, red) / (float)QD + 1e-5f);
  float e = 0.f;
  for (int d = threadIdx.x; d < QD; d += blockDim.x) {
    float o = (yr[d] - mu) * rstd * w[d] + b[d];
    tmp[d] = o;
    e += o * o;
  }
  float energy = sqrtf(qfm_block_sum(e, red));
  float inv = 1.f / (energy + 1e-12f);
  size_t base = (size_t)row * QD;
  for (int d = threadIdx.x; d < QD; d += blockDim.x)
    outb[base + d] = __float2bfloat16(tmp[d] * inv);
}

__global__ void qfm_zero(float* p) {
  if (threadIdx.x == 0 && blockIdx.x == 0) *p = 0.f;
}

// shifted cross-entropy: rows r = b*(S-1)+s -> logits[b,s,:], target labels[b,s+1]
__global__ void __launch_bounds__(256)
qfm_ce(const float* __restrict__ logits, const int* __restrict__ labels,
       float* __restrict__ loss, float invN) {
  __shared__ float red[256];
  int r = blockIdx.x;
  int b = r / (QS - 1), s = r % (QS - 1);
  const float* row = logits + ((size_t)b * QS + s) * QV;
  int tgt = labels[b * QS + s + 1];
  float mx = -3.4e38f;
  for (int v = threadIdx.x; v < QV; v += blockDim.x) mx = fmaxf(mx, row[v]);
  mx = qfm_block_max(mx, red);
  float sum = 0.f;
  for (int v = threadIdx.x; v < QV; v += blockDim.x) sum += expf(row[v] - mx);
  float tot = qfm_block_sum(sum, red);
  if (threadIdx.x == 0)
    atomicAdd(loss, (logf(tot) + mx - row[tgt]) * invN);
}

// ---------------------------------------------------------------------------
// Host launch
// ---------------------------------------------------------------------------
extern "C" void kernel_launch(void* const* d_in, const int* in_sizes, int n_in,
                              void* d_out, int out_size, void* d_ws, size_t ws_size,
                              hipStream_t stream) {
  (void)in_sizes; (void)n_in; (void)out_size; (void)ws_size;
  const int* input_ids = (const int*)d_in[0];
  const int* labels    = (const int*)d_in[1];
  const float* proj_w  = (const float*)d_in[2];
  const float* proj_b  = (const float*)d_in[3];
  const float* ln_w    = (const float*)d_in[4];
  const float* ln_b    = (const float*)d_in[5];
  const float* fln_w   = (const float*)d_in[6];
  const float* fln_b   = (const float*)d_in[7];
  const float* out_w   = (const float*)d_in[8];

  const size_t BSD = (size_t)QB * QS * QD;        // 4,194,304
  const size_t BSS = (size_t)QB * QS * QS;        // 8,388,608
  const size_t PW  = (size_t)QL * QD * QD;        // 8,388,608
  const size_t OW  = (size_t)QD * QV;             // 32,768,000

  char* w = (char*)d_ws;
  size_t off = 0;
  auto carve = [&](size_t bytes) {
    void* p = w + off;
    off = (off + bytes + 255) & ~(size_t)255;
    return p;
  };
  float*          states_f = (float*)carve(BSD * 4);
  __hip_bfloat16* states_b = (__hip_bfloat16*)carve(BSD * 2);
  __hip_bfloat16* midev_b  = (__hip_bfloat16*)carve(BSD * 2);  // mid then ev (bf16)
  float*          k1       = (float*)carve(BSD * 4);           // k1 then ev (f32)
  float*          k2       = (float*)carve(BSD * 4);           // k2 then y  (f32)
  __hip_bfloat16* score_b  = (__hip_bfloat16*)carve(BSS * 2);
  __hip_bfloat16* projw_b  = (__hip_bfloat16*)carve(PW * 2);
  __hip_bfloat16* outw_b   = (__hip_bfloat16*)carve(OW * 2);
  __hip_bfloat16* snorm_b  = (__hip_bfloat16*)carve(BSD * 2);
  float*          energy   = (float*)carve((size_t)QB * QS * 4);
  float*          thr      = (float*)carve((size_t)QB * 4);

  const int ROWS = QB * QS;                    // 4096
  dim3 blk256(256), blk128(128);

  // weight casts (deterministic each call)
  qfm_cast_bf16<<<2048, blk256, 0, stream>>>(proj_w, projw_b, PW);
  qfm_cast_bf16<<<4096, blk256, 0, stream>>>(out_w, outw_b, OW);

  // encode
  qfm_encode<<<ROWS, blk256, 0, stream>>>(input_ids, states_f, states_b);

  dim3 gScore(QS / 128, QS / 64, QB);          // (16,32,2)
  dim3 gFlux(QS / 128, QD / 64, QB);           // (16,16,2)
  dim3 gProj(ROWS / 128, QD / 64, 1);          // (32,16,1)

  for (int i = 0; i < QL; ++i) {
    // k1 = flux(states)
    qfm_gemm_nt_causal<<<gScore, blk128, 0, stream>>>(states_b, score_b, QS, QD, QSCALE);
    qfm_gemm_nn<<<gFlux, blk256, 0, stream>>>(score_b, states_b, k1, nullptr,
        QS, QD, QS, QD, QD, (size_t)QS * QS, (size_t)QS * QD, (size_t)QS * QD);
    // mid = states + dt*k1 (bf16)
    qfm_mid<<<4096, blk256, 0, stream>>>(states_f, k1, midev_b, BSD);
    // k2 = flux(mid)
    qfm_gemm_nt_causal<<<gScore, blk128, 0, stream>>>(midev_b, score_b, QS, QD, QSCALE);
    qfm_gemm_nn<<<gFlux, blk256, 0, stream>>>(score_b, midev_b, k2, nullptr,
        QS, QD, QS, QD, QD, (size_t)QS * QS, (size_t)QS * QD, (size_t)QS * QD);
    // ev = states + 0.5*dt*(k1+k2)  (into k1)
    qfm_heun<<<4096, blk256, 0, stream>>>(states_f, k1, k2, BSD);
    // tunneling
    qfm_energy<<<ROWS, blk256, 0, stream>>>(k1, energy);
    qfm_thr<<<QB, blk256, 0, stream>>>(energy, thr);
    qfm_tunnel<<<ROWS, blk256, 0, stream>>>(k1, energy, thr, midev_b);  // ev_bf16
    // y = ev @ proj_w[i] + proj_b[i]  (into k2)
    qfm_gemm_nn<<<gProj, blk256, 0, stream>>>(midev_b, projw_b + (size_t)i * QD * QD,
        k2, proj_b + (size_t)i * QD, ROWS, QD, QD, QD, QD, 0, 0, 0);
    // states = LN(y)
    qfm_ln<<<ROWS, blk256, 0, stream>>>(k2, ln_w + (size_t)i * QD, ln_b + (size_t)i * QD,
                                        states_f, states_b);
  }

  // final LN + energy normalize -> bf16
  qfm_final<<<ROWS, blk256, 0, stream>>>(states_f, fln_w, fln_b, snorm_b);

  // logits = snorm @ out_w  (f32 into d_out)
  float* logits = (float*)d_out;
  dim3 gLog(ROWS / 128, QV / 64, 1);           // (32,500,1)
  qfm_gemm_nn<<<gLog, blk256, 0, stream>>>(snorm_b, outw_b, logits, nullptr,
      ROWS, QV, QD, QV, QV, 0, 0, 0);

  // loss
  float* lossp = logits + (size_t)QB * QS * QV;
  qfm_zero<<<1, 1, 0, stream>>>(lossp);
  int nrows = QB * (QS - 1);
  qfm_ce<<<nrows, blk256, 0, stream>>>(logits, labels, lossp, 1.f / (float)nrows);
}
